// VQVAEEMAResetMoTokenNetwork_72919954752234
// MI455X (gfx1250) — matrix operand order repaired and checked by reading, not compiled, not run on hardware
//
#include <hip/hip_runtime.h>
#include <hip/hip_bf16.h>
#include <math.h>

typedef __attribute__((ext_vector_type(16))) _Float16 v16h;
typedef __attribute__((ext_vector_type(8)))  float    v8f;
typedef __attribute__((ext_vector_type(4)))  float    v4f;

#define MUF 0.99f
#define ONE_MINUS_MUF 0.01f
#define EPSF 1e-7f
#define DD 512
#define KK 512

// ---------------------------------------------------------------------------
// prep: codebook fp32 -> f16 (workspace), cnorm[k] = ||c_k||^2, zero cs/cc
// ---------------------------------------------------------------------------
__global__ void prep_kernel(const float* __restrict__ cb,
                            _Float16* __restrict__ cb16,
                            float* __restrict__ cnorm,
                            float* __restrict__ cs,
                            float* __restrict__ cc) {
  const int k = blockIdx.x;     // one block per code row, 256 threads
  const int t = threadIdx.x;
  const float* row = cb + (size_t)k * DD;
  float s = 0.f;
  for (int d = t; d < DD; d += 256) {
    float v = row[d];
    cb16[(size_t)k * DD + d] = (_Float16)v;
    cs[(size_t)k * DD + d]   = 0.f;
    s += v * v;
  }
  __shared__ float red[256];
  red[t] = s;
  __syncthreads();
  for (int off = 128; off > 0; off >>= 1) {
    if (t < off) red[t] += red[t + off];
    __syncthreads();
  }
  if (t == 0) cnorm[k] = red[0];
  if (k == 0) {
    for (int d = t; d < KK; d += 256) cc[d] = 0.f;
  }
}

// ---------------------------------------------------------------------------
// main: per-wave 16-row tile. WMMA f16 distance GEMM (B software-pipelined
// through a 4-slot register buffer, prefetch distance 3) + argmin + scatter.
// ---------------------------------------------------------------------------
__launch_bounds__(256, 1)
__global__ void vq_assign_kernel(const float* __restrict__ z,
                                 const float* __restrict__ cb,
                                 const _Float16* __restrict__ cb16,
                                 const float* __restrict__ cnorm,
                                 float* __restrict__ out_bar,
                                 float* __restrict__ out_q,
                                 float* __restrict__ cs,
                                 float* __restrict__ cc) {
  const int lane    = threadIdx.x & 31;
  const int wave    = threadIdx.x >> 5;
  const int rowbase = (blockIdx.x * 8 + wave) * 16;   // 16 rows per wave
  const int mrow = lane & 15;          // A-matrix row held by this lane
  const int kb   = (lane >> 4) * 16;   // A K-half selector (0 or 16)
  const int koff = (lane >> 4) * 16;   // B K-half selector (0 or 16)
  const int col  = lane & 15;          // B column / code within k-tile

  // Convert this wave's 16x512 z-tile to f16 A fragments, held in registers.
  v16h a[16];
  {
    const float* zr = z + (size_t)(rowbase + mrow) * DD;
#pragma unroll
    for (int ks = 0; ks < 16; ++ks) {
      const int d0 = ks * 32 + kb;
      const v4f f0 = *(const v4f*)(zr + d0);
      const v4f f1 = *(const v4f*)(zr + d0 + 4);
      const v4f f2 = *(const v4f*)(zr + d0 + 8);
      const v4f f3 = *(const v4f*)(zr + d0 + 12);
      a[ks][0]  = (_Float16)f0[0]; a[ks][1]  = (_Float16)f0[1];
      a[ks][2]  = (_Float16)f0[2]; a[ks][3]  = (_Float16)f0[3];
      a[ks][4]  = (_Float16)f1[0]; a[ks][5]  = (_Float16)f1[1];
      a[ks][6]  = (_Float16)f1[2]; a[ks][7]  = (_Float16)f1[3];
      a[ks][8]  = (_Float16)f2[0]; a[ks][9]  = (_Float16)f2[1];
      a[ks][10] = (_Float16)f2[2]; a[ks][11] = (_Float16)f2[3];
      a[ks][12] = (_Float16)f3[0]; a[ks][13] = (_Float16)f3[1];
      a[ks][14] = (_Float16)f3[2]; a[ks][15] = (_Float16)f3[3];
    }
  }

  float bestv[8];
  int   besti[8];
#pragma unroll
  for (int i = 0; i < 8; ++i) { bestv[i] = 3.4e38f; besti[i] = 0; }

  // B fragment f = kt*16 + ks lives at cb16[(kt*16+col)*DD + koff + ks*32],
  // rotating through bbuf[f & 3]; fragment f is loaded at step f-3.
  v16h bbuf[4];
  {
    const _Float16* b0 = cb16 + (size_t)col * DD + koff;
    bbuf[0] = *(const v16h*)(b0 + 0 * 32);
    bbuf[1] = *(const v16h*)(b0 + 1 * 32);
    bbuf[2] = *(const v16h*)(b0 + 2 * 32);
  }

  auto tile_step = [&](int kt, int pfmax) {
    v8f c = {};
#pragma unroll
    for (int ks = 0; ks < 16; ++ks) {
      if (ks < pfmax) {                       // prefetch fragment f+3
        const int fn  = kt * 16 + ks + 3;
        const int ktn = fn >> 4;
        const int ksn = fn & 15;
        bbuf[(ks + 3) & 3] =
            *(const v16h*)(cb16 + (size_t)((ktn << 4) + col) * DD + koff +
                           (ksn << 5));
      }
      c = __builtin_amdgcn_wmma_f32_16x16x32_f16(false, a[ks], false,
                                                 bbuf[ks & 3], (short)0, c,
                                                 false, false);
    }
    const float cn  = cnorm[(kt << 4) + col];
    const int   idx = (kt << 4) + col;
#pragma unroll
    for (int i = 0; i < 8; ++i) {
      float s = cn - 2.0f * c[i];          // ||c||^2 - 2 z.c  (||z||^2 drops)
      if (s < bestv[i]) { bestv[i] = s; besti[i] = idx; }
    }
  };

  for (int kt = 0; kt < KK / 16 - 1; ++kt) tile_step(kt, 16);
  tile_step(KK / 16 - 1, 13);              // last tile: only frags < 512 exist

  // Butterfly argmin across the 16 columns (stays within each 16-lane half).
#pragma unroll
  for (int m = 1; m < 16; m <<= 1) {
#pragma unroll
    for (int i = 0; i < 8; ++i) {
      float ov = __shfl_xor(bestv[i], m, 32);
      int   oi = __shfl_xor(besti[i], m, 32);
      if (ov < bestv[i] || (ov == bestv[i] && oi < besti[i])) {
        bestv[i] = ov; besti[i] = oi;
      }
    }
  }

  // Broadcast the 16 winning indices to the whole wave.
  int idxs[16];
#pragma unroll
  for (int r = 0; r < 8; ++r) {
    idxs[r]     = __shfl(besti[r], 0, 32);
    idxs[r + 8] = __shfl(besti[r], 16, 32);
  }

  // Gather codebook rows, write z_q_x and z_q_x_bar (non-temporal: pure
  // streaming stores, keep the codebook hot in L2), segment-sum atomics.
  for (int r = 0; r < 16; ++r) {
    const int idx = idxs[r];
    const v4f* crow = (const v4f*)(cb + (size_t)idx * DD);
    const v4f* zr4  = (const v4f*)(z + (size_t)(rowbase + r) * DD);
    v4f* ob = (v4f*)(out_bar + (size_t)(rowbase + r) * DD);
    v4f* oq = (v4f*)(out_q   + (size_t)(rowbase + r) * DD);
    float* csrow = cs + (size_t)idx * DD;
    for (int j = lane; j < DD / 4; j += 32) {
      v4f qv = crow[j];
      v4f zv = zr4[j];
      v4f bar = zv + (qv - zv);
      __builtin_nontemporal_store(qv,  &oq[j]);
      __builtin_nontemporal_store(bar, &ob[j]);
      atomicAdd(csrow + 4 * j + 0, zv[0]);
      atomicAdd(csrow + 4 * j + 1, zv[1]);
      atomicAdd(csrow + 4 * j + 2, zv[2]);
      atomicAdd(csrow + 4 * j + 3, zv[3]);
    }
    if (lane == 0) atomicAdd(cc + idx, 1.0f);
  }
}

// ---------------------------------------------------------------------------
// finalize 1: new_code_count, perplexity, usage_total (one block of K threads)
// ---------------------------------------------------------------------------
__global__ void finalize_count_kernel(const float* __restrict__ cc_batch,
                                      const float* __restrict__ code_count,
                                      float* __restrict__ out_ncc,
                                      float* __restrict__ out_perp,
                                      float* __restrict__ out_usage) {
  const int t = threadIdx.x;  // K = 512 threads
  float ccb = cc_batch[t];
  float ncc = MUF * code_count[t] + ONE_MINUS_MUF * ccb;
  out_ncc[t] = ncc;
  float usage = (ncc >= 1.0f) ? 1.0f : 0.0f;

  __shared__ float red[KK];
  red[t] = ccb;
  __syncthreads();
  for (int off = KK / 2; off > 0; off >>= 1) {
    if (t < off) red[t] += red[t + off];
    __syncthreads();
  }
  float tot = red[0];
  __syncthreads();
  float p = ccb / tot;
  red[t] = p * logf(p + EPSF);
  __syncthreads();
  for (int off = KK / 2; off > 0; off >>= 1) {
    if (t < off) red[t] += red[t + off];
    __syncthreads();
  }
  float ent = red[0];
  __syncthreads();
  red[t] = usage;
  __syncthreads();
  for (int off = KK / 2; off > 0; off >>= 1) {
    if (t < off) red[t] += red[t + off];
    __syncthreads();
  }
  if (t == 0) {
    out_perp[0]  = expf(-ent);
    out_usage[0] = red[0];
  }
}

// ---------------------------------------------------------------------------
// finalize 2: EMA code_sum + codebook update / reset  (K*D elements)
// ---------------------------------------------------------------------------
__global__ void finalize_update_kernel(const float* __restrict__ code_sum,
                                       const float* __restrict__ cs_batch,
                                       const float* __restrict__ ncc,
                                       const float* __restrict__ z,
                                       float* __restrict__ out_ncs,
                                       float* __restrict__ out_ncb) {
  const size_t i = (size_t)blockIdx.x * blockDim.x + threadIdx.x;
  const int k = (int)(i >> 9);  // / DD
  float ncs = MUF * code_sum[i] + ONE_MINUS_MUF * cs_batch[i];
  __builtin_nontemporal_store(ncs, &out_ncs[i]);
  float c = ncc[k];
  float ncb = (c >= 1.0f) ? (ncs / c) : z[i];  // code_rand = z_e_x[:K]
  __builtin_nontemporal_store(ncb, &out_ncb[i]);
}

// ---------------------------------------------------------------------------
extern "C" void kernel_launch(void* const* d_in, const int* in_sizes, int n_in,
                              void* d_out, int out_size, void* d_ws, size_t ws_size,
                              hipStream_t stream) {
  const float* z          = (const float*)d_in[0];  // (N, D) fp32
  const float* cb         = (const float*)d_in[1];  // (K, D) fp32
  const float* code_sum   = (const float*)d_in[2];  // (K, D) fp32
  const float* code_count = (const float*)d_in[3];  // (K,)   fp32
  const int N = in_sizes[0] / DD;

  // Workspace: f16 codebook (512KB) | cnorm (2KB) | cc (2KB) | cs (1MB)
  char* w = (char*)d_ws;
  _Float16* cb16 = (_Float16*)w;
  float* cnorm   = (float*)(w + (size_t)KK * DD * 2);
  float* cc_b    = (float*)(w + (size_t)KK * DD * 2 + 4096);
  float* cs_b    = (float*)(w + (size_t)KK * DD * 2 + 8192);

  // Outputs, flat in return order.
  float* out       = (float*)d_out;
  float* out_bar   = out;                           // (N, D)
  float* out_q     = out + (size_t)N * DD;          // (N, D)
  float* out_perp  = out + 2 * (size_t)N * DD;      // scalar
  float* out_usage = out_perp + 1;                  // scalar
  float* out_ncb   = out_perp + 2;                  // (K, D)
  float* out_ncs   = out_ncb + (size_t)KK * DD;     // (K, D)
  float* out_ncc   = out_ncs + (size_t)KK * DD;     // (K,)

  prep_kernel<<<KK, 256, 0, stream>>>(cb, cb16, cnorm, cs_b, cc_b);
  vq_assign_kernel<<<N / 128, 256, 0, stream>>>(z, cb, cb16, cnorm,
                                                out_bar, out_q, cs_b, cc_b);
  finalize_count_kernel<<<1, KK, 0, stream>>>(cc_b, code_count,
                                              out_ncc, out_perp, out_usage);
  finalize_update_kernel<<<(KK * DD) / 256, 256, 0, stream>>>(code_sum, cs_b,
                                                              out_ncc, z,
                                                              out_ncs, out_ncb);
}
